// VocosISTFTHead_26929444946253
// MI455X (gfx1250) — matrix operand order repaired and checked by reading, not compiled, not run on hardware
//
#include <hip/hip_runtime.h>
#include <cstdint>

typedef __attribute__((ext_vector_type(16))) _Float16 v16h;
typedef __attribute__((ext_vector_type(8)))  _Float16 v8h;
typedef __attribute__((ext_vector_type(4)))  _Float16 v4h;
typedef __attribute__((ext_vector_type(8)))  float    v8f;
typedef __attribute__((ext_vector_type(4)))  float    v4f;
typedef __attribute__((ext_vector_type(4)))  int      v4i;

#define B_SZ   16
#define T_SZ   2048
#define H_SZ   512
#define NCH    1026      // N_FFT + 2
#define HALF   513
#define KP     1056      // NCH padded to multiple of 32
#define NP     1152      // NCH padded to multiple of 128 (spec/Wt column pad)
#define NFFT   1024
#define HOP    256
#define PADW   384
#define OUTL   524544    // (T-1)*HOP + WIN - 2*PAD + HOP
#define MTOT   (B_SZ * T_SZ)   // 32768

#define LDA 40    // 32 + 8 pad (halves)
#define LDB 136   // 128 + 8 pad (halves)

// ---- 16B global -> LDS copy: async (ASYNCcnt) when available, else via VGPRs ----
__device__ __forceinline__ void cp16(_Float16* lds_dst, const _Float16* g_src) {
#if __has_builtin(__builtin_amdgcn_global_load_async_to_lds_b128)
    __builtin_amdgcn_global_load_async_to_lds_b128(
        (__attribute__((address_space(1))) v4i*)(uintptr_t)g_src,
        (__attribute__((address_space(3))) v4i*)lds_dst, 0, 0);
#else
    *(v8h*)lds_dst = *(const v8h*)g_src;
#endif
}

__device__ __forceinline__ void cp_wait_barrier() {
#if __has_builtin(__builtin_amdgcn_global_load_async_to_lds_b128)
  #if __has_builtin(__builtin_amdgcn_s_wait_asynccnt)
    __builtin_amdgcn_s_wait_asynccnt(0);
  #else
    asm volatile("s_wait_asynccnt 0x0" ::: "memory");
  #endif
#endif
    __syncthreads();
}

// ---- fragment loaders from LDS (CDNA5 wave32 WMMA layouts, ISA 7.12.2) ----
__device__ __forceinline__ v16h load_frag_a(const _Float16* As, int lane, int mbase) {
    int m    = mbase + (lane & 15);
    int ktop = (lane >> 4) * 8;        // lanes 0-15: K 0..7/16..23, lanes 16-31: K 8..15/24..31
    const _Float16* p = As + m * LDA + ktop;
    v8h lo = *(const v8h*)(p);
    v8h hi = *(const v8h*)(p + 16);
    return __builtin_shufflevector(lo, hi, 0,1,2,3,4,5,6,7,8,9,10,11,12,13,14,15);
}

__device__ __forceinline__ v16h load_frag_b(const _Float16* Bs, int lane, int nbase) {
    const int n    = nbase + (lane & 15);
    const int ktop = (lane >> 4) * 8;   // lane = N column; K 0..7/16..23 or 8..15/24..31
    v16h r;
#pragma unroll
    for (int s = 0; s < 8; ++s) {
        r[s]     = Bs[(ktop + s) * LDB + n];
        r[s + 8] = Bs[(16 + ktop + s) * LDB + n];
    }
    return r;
}

// ---- unified f16 WMMA GEMM: C[m,n] = sum_k A[m,k]*B[k,n] (+bias[n]) ----
// Block 128x128, 8 waves (4x2), wave tile 32x64, double-buffered async LDS staging.
template <bool HAS_BIAS>
__global__ __launch_bounds__(256) void vocos_gemm(
    const _Float16* __restrict__ Ag, int lda,
    const _Float16* __restrict__ Bg, int ldb,
    float* __restrict__ Cg, int ldc,
    const float* __restrict__ biasp, int K) {
    __shared__ _Float16 As[2][128 * LDA];
    __shared__ _Float16 Bs[2][32 * LDB];
    const int tid  = threadIdx.x;
    const int lane = tid & 31, wave = tid >> 5;
    const int wr = wave >> 1, wc = wave & 1;
    const int m0 = blockIdx.x * 128;
    const int n0 = blockIdx.y * 128;

    v8f acc[2][4];
#pragma unroll
    for (int i = 0; i < 2; ++i)
#pragma unroll
        for (int j = 0; j < 4; ++j) acc[i][j] = {};

    // per-thread copy coordinates (2 x 16B for A tile, 2 x 16B for B tile)
    const int ar0 = tid >> 2,         ak0 = (tid & 3) << 3;
    const int ar1 = (tid + 256) >> 2, ak1 = ((tid + 256) & 3) << 3;
    const int bk0 = tid >> 4,         bn0 = (tid & 15) << 3;
    const int bk1 = (tid + 256) >> 4, bn1 = ((tid + 256) & 15) << 3;

    auto load_tiles = [&](int k0, int buf) {
        cp16(&As[buf][ar0 * LDA + ak0], Ag + (size_t)(m0 + ar0) * lda + k0 + ak0);
        cp16(&As[buf][ar1 * LDA + ak1], Ag + (size_t)(m0 + ar1) * lda + k0 + ak1);
        cp16(&Bs[buf][bk0 * LDB + bn0], Bg + (size_t)(k0 + bk0) * ldb + n0 + bn0);
        cp16(&Bs[buf][bk1 * LDB + bn1], Bg + (size_t)(k0 + bk1) * ldb + n0 + bn1);
    };

    const int nk = K >> 5;
    load_tiles(0, 0);
    for (int i = 0; i < nk; ++i) {
        const int buf = i & 1;
        cp_wait_barrier();                 // tile i resident; all waves done with buf^1
        if (i + 1 < nk) load_tiles((i + 1) << 5, buf ^ 1);

        v16h af[2], bf[4];
#pragma unroll
        for (int ii = 0; ii < 2; ++ii) af[ii] = load_frag_a(As[buf], lane, wr * 32 + ii * 16);
#pragma unroll
        for (int j = 0; j < 4; ++j)   bf[j]  = load_frag_b(Bs[buf], lane, wc * 64 + j * 16);
#pragma unroll
        for (int ii = 0; ii < 2; ++ii)
#pragma unroll
            for (int j = 0; j < 4; ++j)
                acc[ii][j] = __builtin_amdgcn_wmma_f32_16x16x32_f16(
                    false, af[ii], false, bf[j], (short)0, acc[ii][j], false, false);
    }

    const int lrow = (lane >> 4) * 8;
    const int lcol = lane & 15;
#pragma unroll
    for (int i = 0; i < 2; ++i)
#pragma unroll
        for (int j = 0; j < 4; ++j) {
            int n  = n0 + wc * 64 + j * 16 + lcol;
            int mb = m0 + wr * 32 + i * 16 + lrow;
            float bv = HAS_BIAS ? biasp[n] : 0.f;
#pragma unroll
            for (int r = 0; r < 8; ++r)
                __builtin_nontemporal_store(acc[i][j][r] + bv,
                                            &Cg[(size_t)(mb + r) * ldc + n]);
        }
}

// ---- prep: hidden f32 -> f16 (one pass; GEMM1 then re-reads half the bytes) ----
__global__ __launch_bounds__(256) void vocos_prep_h16(
    const float* __restrict__ hid, _Float16* __restrict__ h16) {
    size_t idx = (size_t)(blockIdx.x * 256 + threadIdx.x) * 4;   // MTOT*H_SZ elems
    v4f f = *(const v4f*)(hid + idx);
    *(v4h*)(h16 + idx) = __builtin_convertvector(f, v4h);
}

// ---- prep: Wt[k, c] = W[c, k] in f16, zero-padded to NP columns; padded bias ----
__global__ __launch_bounds__(256) void vocos_prep_wt(
    const float* __restrict__ W, const float* __restrict__ bias,
    _Float16* __restrict__ Wt, float* __restrict__ biasp) {
    int idx = blockIdx.x * 256 + threadIdx.x;   // H_SZ * NP threads
    int k = idx / NP;
    int c = idx - k * NP;
    float v = (c < NCH) ? W[(size_t)c * H_SZ + k] : 0.f;
    Wt[idx] = (_Float16)v;
    if (k == 0) biasp[c] = (c < NCH) ? bias[c] : 0.f;
}

// ---- elementwise: q[t,k]=mag*cos(ph), q[t,k+513]=mag*sin(ph); zero K-pad ----
__global__ __launch_bounds__(256) void vocos_build_q(
    const float* __restrict__ spec, _Float16* __restrict__ q) {
    int idx = blockIdx.x * 256 + threadIdx.x;   // MTOT * 528 threads
    int t  = idx / 528;
    int kk = idx - t * 528;
    const float* sp = spec + (size_t)t * NP;
    _Float16* qp = q + (size_t)t * KP;
    if (kk < HALF) {
        float mag = fminf(__expf(sp[kk]), 100.f);
        float ph  = sp[kk + HALF];
        qp[kk]        = (_Float16)(mag * __cosf(ph));
        qp[kk + HALF] = (_Float16)(mag * __sinf(ph));
    } else {
        int z = kk - HALF;                       // 15 threads cover 30 pad halves
        qp[NCH + z]      = (_Float16)0.f;
        qp[NCH + 15 + z] = (_Float16)0.f;
    }
}

// ---- basis: U[k,n] = a_k * window[n] * {cos | -sin}(2*pi*k*n/1024), exact mod-1024 ----
__global__ __launch_bounds__(256) void vocos_build_basis(
    const float* __restrict__ win, _Float16* __restrict__ U) {
    int idx = blockIdx.x * 256 + threadIdx.x;   // KP * NFFT threads
    int k = idx >> 10;
    int n = idx & (NFFT - 1);
    float val = 0.f;
    if (k < NCH) {
        int kk = (k < HALF) ? k : (k - HALF);
        float a = (kk == 0 || kk == NFFT / 2) ? (1.f / NFFT) : (2.f / NFFT);
        int p = (kk * n) & (NFFT - 1);
        float th = 6.283185307179586f * (float)p * (1.f / NFFT);
        float tr = (k < HALF) ? __cosf(th) : -__sinf(th);
        val = a * tr * win[n];
    }
    U[(size_t)k * NFFT + n] = (_Float16)val;
}

// ---- overlap-add gather (<=4 contributions/sample) + window^2 normalization ----
__global__ __launch_bounds__(256) void vocos_ola(
    const float* __restrict__ frames, const float* __restrict__ win,
    float* __restrict__ out, int total) {
    int idx = blockIdx.x * 256 + threadIdx.x;
    if (idx >= total) return;
    int b  = idx / OUTL;
    int lp = idx - b * OUTL;
    int l  = lp + PADW;
    int tmax = l >> 8;                              if (tmax > T_SZ - 1) tmax = T_SZ - 1;
    int tmin = (l - (NFFT - 1) + (HOP - 1)) >> 8;   if (tmin < 0) tmin = 0;
    float s = 0.f, nrm = 0.f;
    for (int t = tmin; t <= tmax; ++t) {
        int n = l - (t << 8);
        s += frames[((size_t)(b * T_SZ + t)) * NFFT + n];
        float w = win[n];
        nrm += w * w;
    }
    out[idx] = s / nrm;
}

extern "C" void kernel_launch(void* const* d_in, const int* in_sizes, int n_in,
                              void* d_out, int out_size, void* d_ws, size_t ws_size,
                              hipStream_t stream) {
    const float* hidden = (const float*)d_in[0];   // (16, 2048, 512) f32
    const float* W      = (const float*)d_in[1];   // (1026, 512) f32
    const float* bias   = (const float*)d_in[2];   // (1026,) f32
    const float* window = (const float*)d_in[3];   // (1024,) f32
    float* out = (float*)d_out;

    char* ws = (char*)d_ws;
    const size_t SPEC_B = (size_t)MTOT * NP * sizeof(float);       // 150,994,944
    const size_t Q_B    = (size_t)MTOT * KP * sizeof(_Float16);    //  69,206,016
    const size_t U_B    = (size_t)KP * NFFT * sizeof(_Float16);    //   2,162,688
    const size_t H16_B  = (size_t)MTOT * H_SZ * sizeof(_Float16);  //  33,554,432
    const size_t WT_B   = (size_t)H_SZ * NP * sizeof(_Float16);    //   1,179,648

    float*     spec   = (float*)ws;                 // stride NP
    float*     frames = (float*)ws;                 // aliases spec (spec dead before GEMM2)
    _Float16*  q      = (_Float16*)(ws + SPEC_B);
    _Float16*  U      = (_Float16*)(ws + SPEC_B + Q_B);
    _Float16*  h16    = (_Float16*)(ws + SPEC_B + Q_B + U_B);
    _Float16*  Wt     = (_Float16*)(ws + SPEC_B + Q_B + U_B + H16_B);
    float*     biasp  = (float*)(ws + SPEC_B + Q_B + U_B + H16_B + WT_B);

    vocos_prep_h16<<<(MTOT * H_SZ / 4) / 256, 256, 0, stream>>>(hidden, h16);
    vocos_prep_wt<<<(H_SZ * NP) / 256, 256, 0, stream>>>(W, bias, Wt, biasp);
    vocos_build_basis<<<(KP * NFFT) / 256, 256, 0, stream>>>(window, U);

    // GEMM1: spec[m, c] = h16[m, :] . Wt[:, c] + bias[c]   (M=32768, N=1152, K=512)
    vocos_gemm<true><<<dim3(MTOT / 128, NP / 128), 256, 0, stream>>>(
        h16, H_SZ, Wt, NP, spec, NP, biasp, H_SZ);

    vocos_build_q<<<(MTOT * 528) / 256, 256, 0, stream>>>(spec, q);

    // GEMM2: frames[m, n] = q[m, :] . U[:, n]              (M=32768, N=1024, K=1056)
    vocos_gemm<false><<<dim3(MTOT / 128, NFFT / 128), 256, 0, stream>>>(
        q, KP, U, NFFT, frames, NFFT, nullptr, KP);

    int total = B_SZ * OUTL;
    vocos_ola<<<(total + 255) / 256, 256, 0, stream>>>(frames, window, out, total);
}
